// JCViT_46815143527106
// MI455X (gfx1250) — compile-verified, matrix operand order checked by reading, not run on hardware
//
#include <hip/hip_runtime.h>
#include <hip/hip_bf16.h>

// ---------------- CDNA5 WMMA types ----------------
typedef __attribute__((ext_vector_type(16))) _Float16 v16h;
typedef __attribute__((ext_vector_type(8)))  float    v8f;
typedef __attribute__((ext_vector_type(8)))  _Float16 half8;

// ViT-base/16 constants
#define VD   768
#define VNH  12
#define VDH  64
#define VL   12
#define VT   197
#define VTP  208              // padded T (16B-aligned f16 stride)
#define VN   8
#define VROWS (VN * VT)       // 1576

static __device__ __forceinline__ int imin(int a, int b) { return a < b ? a : b; }
static __device__ __forceinline__ int imax(int a, int b) { return a > b ? a : b; }

// =====================================================================
// Batched WMMA GEMM:  Out = act(A * B^T + bias) [+ Res]
//   A : M x K row-major f16 (lda);  base = A + outer*aO + inner*aI
//   Bt: Ncols x K row-major f16 (ldb) -- weights pre-transposed to [n][k]
//   C : off = outer*cO + inner*cI + m*ldc + n
// 128 threads = 4 wave32; block tile 64x64; each wave a 32x32 quadrant
// (2x2 WMMA 16x16 tiles, K-step 32 -> 4 v_wmma per wave per barrier).
// Fast path: global_load_b128 -> ds_store_b128 staging.
// =====================================================================
template<int ACT, bool HAS_RES, bool WRITE_H>
__global__ __launch_bounds__(128)
void gemm_wmma_kernel(const _Float16* __restrict__ A, const _Float16* __restrict__ Bt,
                      const float* __restrict__ bias, const float* __restrict__ Res,
                      float* __restrict__ Out, _Float16* __restrict__ OutH,
                      int M, int Ncols, int K, int lda, int ldb, int ldc,
                      int innerCnt,
                      long aO, long aI, long bO, long bI, long cO, long cI)
{
    __shared__ __align__(16) _Float16 As[64][40];   // [m][k], 80B row stride
    __shared__ __align__(16) _Float16 Bs[64][40];   // [n][k]

    const int bz    = blockIdx.z;
    const int outer = bz / innerCnt;
    const int inner = bz % innerCnt;
    const _Float16* Ab = A  + outer * aO + inner * aI;
    const _Float16* Bb = Bt + outer * bO + inner * bI;
    const long cBase   = outer * cO + inner * cI;

    const int rowBase = blockIdx.y * 64;
    const int colBase = blockIdx.x * 64;
    const int tid  = threadIdx.x;
    const int lane = tid & 31;
    const int wave = tid >> 5;
    const int wm   = wave >> 1, wn = wave & 1;  // wave quadrant in 64x64 tile
    const int half = lane >> 4, idx = lane & 15;

    const bool aVec = ((lda & 7) == 0) && ((((size_t)Ab) & 15) == 0) && (rowBase + 63 < M);
    const bool bVec = ((ldb & 7) == 0) && ((((size_t)Bb) & 15) == 0) && (colBase + 63 < Ncols);

    v8f acc[2][2] = {};

    for (int k0 = 0; k0 < K; k0 += 32) {
        const bool kFull = (k0 + 32 <= K);
        // ---------------- stage A (64 x 32 f16) ----------------
        if (aVec && kFull) {
            #pragma unroll
            for (int e = 0; e < 2; ++e) {
                int c = tid + e * 128;          // 256 chunks of 8 halves
                int r = c >> 2, sg = (c & 3) * 8;
                *(half8*)&As[r][sg] =
                    *(const half8*)(Ab + (long)(rowBase + r) * lda + k0 + sg);
            }
        } else {
            #pragma unroll
            for (int e = 0; e < 16; ++e) {
                int c = tid + e * 128;          // 2048 elements
                int r = c >> 5, k = c & 31;
                int gm = rowBase + r, gk = k0 + k;
                _Float16 v = (_Float16)0.f;
                if (gm < M && gk < K) v = Ab[(long)gm * lda + gk];
                As[r][k] = v;
            }
        }
        // ---------------- stage B (64 x 32 f16) ----------------
        if (bVec && kFull) {
            #pragma unroll
            for (int e = 0; e < 2; ++e) {
                int c = tid + e * 128;
                int r = c >> 2, sg = (c & 3) * 8;
                *(half8*)&Bs[r][sg] =
                    *(const half8*)(Bb + (long)(colBase + r) * ldb + k0 + sg);
            }
        } else {
            #pragma unroll
            for (int e = 0; e < 16; ++e) {
                int c = tid + e * 128;
                int r = c >> 5, k = c & 31;
                int gn = colBase + r, gk = k0 + k;
                _Float16 v = (_Float16)0.f;
                if (gn < Ncols && gk < K) v = Bb[(long)gn * ldb + gk];
                Bs[r][k] = v;
            }
        }
        // prefetch next K tile (gfx1250 global_prefetch)
        if (k0 + 32 < K) {
            __builtin_prefetch(Ab + (long)imin(rowBase + (tid >> 1), M - 1) * lda + k0 + 32, 0, 1);
            __builtin_prefetch(Bb + (long)imin(colBase + (tid >> 1), Ncols - 1) * ldb + k0 + 32, 0, 1);
        }
        __syncthreads();

        // ------- fragments per documented 16-bit A/B VGPR layouts -------
        v16h af[2], bf[2];
        #pragma unroll
        for (int j = 0; j < 8; ++j) {
            int klo = (half ? 8 : 0) + ((j >= 4) ? 16 : 0) + 2 * (j & 3);
            #pragma unroll
            for (int i = 0; i < 2; ++i) {
                af[i][2 * j]     = As[wm * 32 + i * 16 + idx][klo];
                af[i][2 * j + 1] = As[wm * 32 + i * 16 + idx][klo + 1];
                bf[i][2 * j]     = Bs[wn * 32 + i * 16 + idx][klo];
                bf[i][2 * j + 1] = Bs[wn * 32 + i * 16 + idx][klo + 1];
            }
        }
        #pragma unroll
        for (int i = 0; i < 2; ++i)
            #pragma unroll
            for (int j = 0; j < 2; ++j)
                acc[i][j] = __builtin_amdgcn_wmma_f32_16x16x32_f16(
                                false, af[i], false, bf[j], (short)0, acc[i][j],
                                false, false);
        __syncthreads();
    }

    // ------- epilogue: C/D layout VGPR r -> M = r + (half?8:0), N = idx -------
    #pragma unroll
    for (int ti = 0; ti < 2; ++ti)
        #pragma unroll
        for (int tj = 0; tj < 2; ++tj)
            #pragma unroll
            for (int r = 0; r < 8; ++r) {
                int m = rowBase + wm * 32 + ti * 16 + r + (half ? 8 : 0);
                int n = colBase + wn * 32 + tj * 16 + idx;
                if (m < M && n < Ncols) {
                    float v = acc[ti][tj][r];
                    if (bias) v += bias[n];
                    if (ACT == 1) v = 0.5f * v * (1.f + erff(v * 0.70710678118f)); // GELU
                    else if (ACT == 2) v = fmaxf(v, 0.f);                           // ReLU
                    long off = cBase + (long)m * ldc + n;
                    if (HAS_RES) v += Res[off];
                    Out[off] = v;
                    if (WRITE_H) OutH[off] = (_Float16)v;
                }
            }
}

// =====================================================================
// Weight transpose + f32->f16 convert:  W[K][N] -> Wt[N][K]
// =====================================================================
__global__ __launch_bounds__(256)
void transpose_f16_kernel(const float* __restrict__ W, _Float16* __restrict__ Wt,
                          int K, int N)
{
    __shared__ float tile[32][33];
    int kb = blockIdx.y * 32, nb = blockIdx.x * 32;
    for (int i = threadIdx.y; i < 32; i += 8) {
        int k = kb + i, n = nb + threadIdx.x;
        tile[i][threadIdx.x] = (k < K && n < N) ? W[(long)k * N + n] : 0.f;
    }
    __syncthreads();
    for (int i = threadIdx.y; i < 32; i += 8) {
        int n = nb + i, k = kb + threadIdx.x;
        if (n < N && k < K) Wt[(long)n * K + k] = (_Float16)tile[threadIdx.x][i];
    }
}

// ---- plain f32 -> f16 convert (patch_w already in [n][k] form) ----
__global__ __launch_bounds__(256)
void convert_f16_kernel(const float* __restrict__ W, _Float16* __restrict__ Wt, long n)
{
    long idx = (long)blockIdx.x * 256 + threadIdx.x;
    if (idx < n) Wt[idx] = (_Float16)W[idx];
}

// ---- repack V into V^T [n][h][dh][VTP] f16 (zero-padded t) ----
__global__ __launch_bounds__(256)
void repack_v_kernel(const _Float16* __restrict__ qkv, _Float16* __restrict__ vT)
{
    long idx = (long)blockIdx.x * 256 + threadIdx.x;
    if (idx >= (long)VN * VNH * VDH * VTP) return;
    int t = (int)(idx % VTP);
    long r = idx / VTP;
    int dh = (int)(r % VDH);
    long r2 = r / VDH;
    int h = (int)(r2 % VNH);
    int n = (int)(r2 / VNH);
    _Float16 v = (_Float16)0.f;
    if (t < VT) v = qkv[((long)n * VT + t) * 3 * VD + 2 * VD + h * VDH + dh];
    vT[idx] = v;
}

// =====================================================================
// LayerNorm (eps=1e-6, biased var) -> f16 output
// =====================================================================
__global__ __launch_bounds__(256)
void ln_f16_kernel(const float* __restrict__ in, long inStride,
                   const float* __restrict__ w, const float* __restrict__ b,
                   _Float16* __restrict__ out, long outStride, int rows)
{
    int row = blockIdx.x;
    if (row >= rows) return;
    const float* x = in + (long)row * inStride;
    __shared__ float s1[256], s2[256];
    float sum = 0.f, sum2 = 0.f;
    for (int i = threadIdx.x; i < VD; i += 256) { float v = x[i]; sum += v; sum2 += v * v; }
    s1[threadIdx.x] = sum; s2[threadIdx.x] = sum2;
    __syncthreads();
    for (int off = 128; off > 0; off >>= 1) {
        if (threadIdx.x < off) { s1[threadIdx.x] += s1[threadIdx.x + off];
                                 s2[threadIdx.x] += s2[threadIdx.x + off]; }
        __syncthreads();
    }
    float mu  = s1[0] * (1.f / VD);
    float var = s2[0] * (1.f / VD) - mu * mu;
    float rs  = rsqrtf(var + 1e-6f);
    _Float16* o = out + (long)row * outStride;
    for (int i = threadIdx.x; i < VD; i += 256)
        o[i] = (_Float16)((x[i] - mu) * rs * w[i] + b[i]);
}

// =====================================================================
// Row softmax with scale; f32 in (ld=cols) -> f16 out (ld=outLd)
// =====================================================================
__global__ __launch_bounds__(256)
void softmax_f16_kernel(const float* __restrict__ in, _Float16* __restrict__ out,
                        int cols, int outLd, float scale)
{
    long row = blockIdx.x;
    const float* x = in + row * cols;
    __shared__ float red[256];
    int tid = threadIdx.x;
    float v = (tid < cols) ? x[tid] * scale : -3.0e38f;
    red[tid] = v; __syncthreads();
    for (int off = 128; off > 0; off >>= 1) {
        if (tid < off) red[tid] = fmaxf(red[tid], red[tid + off]);
        __syncthreads();
    }
    float mx = red[0]; __syncthreads();
    float e = (tid < cols) ? expf(v - mx) : 0.f;
    red[tid] = e; __syncthreads();
    for (int off = 128; off > 0; off >>= 1) {
        if (tid < off) red[tid] += red[tid + off];
        __syncthreads();
    }
    float s = red[0];
    if (tid < cols) out[row * outLd + tid] = (_Float16)(e / s);
}

// =====================================================================
// Preprocess: argsort(noise), crop, antialiased triangle resize 72->56,
// jitter+clamp, per-channel unbiased normalize, shuffled 224x224 write.
// =====================================================================
__global__ __launch_bounds__(256)
void preprocess_kernel(const float* __restrict__ x, const float* __restrict__ noise,
                       const int* __restrict__ crop_off, const int* __restrict__ jitter,
                       float* __restrict__ img)
{
    int n    = blockIdx.x >> 4;
    int slot = blockIdx.x & 15;
    __shared__ int   shuf[16];
    __shared__ float pix[3136];
    __shared__ float red0[256], red1[256];
    int tid = threadIdx.x;

    if (tid < 16) {
        float nv = noise[n * 16 + tid];
        int rank = 0;
        for (int k = 0; k < 16; ++k) {
            float ov = noise[n * 16 + k];
            if (ov < nv || (ov == nv && k < tid)) rank++;
        }
        shuf[rank] = tid;   // ids_shuffle
    }
    __syncthreads();
    int j  = shuf[slot];
    int ga = j >> 2, gb = j & 3;
    int o0 = crop_off[(n * 16 + j) * 2 + 0];
    int o1 = crop_off[(n * 16 + j) * 2 + 1];
    int R  = slot >> 2, C = slot & 3;
    const float s = 72.f / 56.f;

    for (int c = 0; c < 3; ++c) {
        float jv = (float)(jitter[(n * 16 + j) * 3 + c] - 2);
        const float* src = x + ((long)n * 3 + c) * 368 * 368;
        float sum = 0.f, sum2 = 0.f;
        for (int p = tid; p < 3136; p += 256) {
            int oy = p / 56, ox = p % 56;
            float cy = (oy + 0.5f) * s - 0.5f;
            float cx = (ox + 0.5f) * s - 0.5f;
            int y0 = (int)ceilf(cy - s), y1 = (int)floorf(cy + s);
            int x0 = (int)ceilf(cx - s), x1 = (int)floorf(cx + s);
            float val = 0.f, wsum = 0.f;
            for (int yy = y0; yy <= y1; ++yy) {
                float wy = 1.f - fabsf((float)yy - cy) / s;
                if (wy <= 0.f) continue;
                int sy = imin(imax(yy, 0), 71);
                for (int xx = x0; xx <= x1; ++xx) {
                    float wx = 1.f - fabsf((float)xx - cx) / s;
                    if (wx <= 0.f) continue;
                    int sx = imin(imax(xx, 0), 71);
                    float pv = src[(long)(ga * 98 + o0 + sy) * 368 + (gb * 98 + o1 + sx)];
                    val += wy * wx * pv; wsum += wy * wx;
                }
            }
            val = val / wsum + jv;
            val = fminf(fmaxf(val, 0.f), 255.f);
            pix[p] = val; sum += val; sum2 += val * val;
        }
        red0[tid] = sum; red1[tid] = sum2;
        __syncthreads();
        for (int off = 128; off > 0; off >>= 1) {
            if (tid < off) { red0[tid] += red0[tid + off]; red1[tid] += red1[tid + off]; }
            __syncthreads();
        }
        float m   = red0[0] * (1.f / 3136.f);
        float var = (red1[0] - red0[0] * red0[0] * (1.f / 3136.f)) * (1.f / 3135.f);
        float sd  = sqrtf(fmaxf(var, 0.f));
        if (sd == 0.f) sd = 1.f;
        for (int p = tid; p < 3136; p += 256) {
            int oy = p / 56, ox = p % 56;
            img[(((long)n * 3 + c) * 224 + (R * 56 + oy)) * 224 + (C * 56 + ox)] =
                (pix[p] - m) / sd;
        }
        __syncthreads();
    }
}

// ---- img -> f16 patch matrix (1568 x 768), feature = c*256 + r*16 + s ----
__global__ __launch_bounds__(256)
void patchify_kernel(const float* __restrict__ img, _Float16* __restrict__ patches)
{
    long idx = (long)blockIdx.x * 256 + threadIdx.x;
    if (idx >= (long)VN * 196 * VD) return;
    int d = (int)(idx % VD);
    long rp = idx / VD;
    int p = (int)(rp % 196);
    int n = (int)(rp / 196);
    int c = d >> 8, r = (d >> 4) & 15, sc = d & 15;
    int ph = p / 14, pw = p % 14;
    patches[idx] = (_Float16)img[(((long)n * 3 + c) * 224 + ph * 16 + r) * 224 + pw * 16 + sc];
}

// ---- assemble tokens: cls + patch-embed + pos ----
__global__ __launch_bounds__(256)
void assemble_tokens_kernel(const float* __restrict__ pe, const float* __restrict__ cls,
                            const float* __restrict__ pos, float* __restrict__ tok)
{
    long idx = (long)blockIdx.x * 256 + threadIdx.x;
    if (idx >= (long)VN * VT * VD) return;
    int d = (int)(idx % VD);
    long rt = idx / VD;
    int t = (int)(rt % VT);
    int n = (int)(rt / VT);
    float v = (t == 0) ? cls[d] : pe[((long)n * 196 + (t - 1)) * VD + d];
    tok[idx] = v + pos[(long)t * VD + d];
}

// ---- target = map_coord[ids_restore] ----
__global__ void target_kernel(const float* __restrict__ noise, float* __restrict__ out)
{
    int t = threadIdx.x;
    if (t >= 128) return;
    int n = t >> 4, i = t & 15;
    float nv = noise[n * 16 + i];
    int rank = 0;
    for (int k = 0; k < 16; ++k) {
        float ov = noise[n * 16 + k];
        if (ov < nv || (ov == nv && k < i)) rank++;
    }
    out[(n * 16 + i) * 2 + 0] = (float)(rank >> 2);
    out[(n * 16 + i) * 2 + 1] = (float)(rank & 3);
}

// =====================================================================
extern "C" void kernel_launch(void* const* d_in, const int* in_sizes, int n_in,
                              void* d_out, int out_size, void* d_ws, size_t ws_size,
                              hipStream_t stream)
{
    const float* x        = (const float*)d_in[0];
    const float* noise    = (const float*)d_in[1];
    const int*   crop_off = (const int*)  d_in[2];
    const int*   jitter   = (const int*)  d_in[3];
    const float* patch_w  = (const float*)d_in[4];
    const float* patch_b  = (const float*)d_in[5];
    const float* cls_tok  = (const float*)d_in[6];
    const float* pos_emb  = (const float*)d_in[7];
    const float* ln1_w    = (const float*)d_in[8];
    const float* ln1_b    = (const float*)d_in[9];
    const float* qkv_w    = (const float*)d_in[10];
    const float* qkv_b    = (const float*)d_in[11];
    const float* proj_w   = (const float*)d_in[12];
    const float* proj_b   = (const float*)d_in[13];
    const float* ln2_w    = (const float*)d_in[14];
    const float* ln2_b    = (const float*)d_in[15];
    const float* mlp_w1   = (const float*)d_in[16];
    const float* mlp_b1   = (const float*)d_in[17];
    const float* mlp_w2   = (const float*)d_in[18];
    const float* mlp_b2   = (const float*)d_in[19];
    const float* lnf_w    = (const float*)d_in[20];
    const float* lnf_b    = (const float*)d_in[21];
    const float* head_w   = (const float*)d_in[22];
    const float* head_b   = (const float*)d_in[23];
    const float* fc1_w    = (const float*)d_in[24];
    const float* fc1_b    = (const float*)d_in[25];
    const float* fc2_w    = (const float*)d_in[26];
    const float* fc2_b    = (const float*)d_in[27];
    float* out = (float*)d_out;

    // ---- carve workspace (~81 MB) ----
    char* wp = (char*)d_ws;
    auto carve = [&](size_t bytes) -> char* {
        char* r = wp; wp += (bytes + 255) & ~(size_t)255; return r;
    };
    float*     img      = (float*)    carve((size_t)VN * 3 * 224 * 224 * 4);
    _Float16*  patches  = (_Float16*) carve((size_t)VN * 196 * VD * 2);
    float*     tokF32   = (float*)    carve((size_t)VROWS * VD * 4);
    _Float16*  aF16     = (_Float16*) carve((size_t)VROWS * VD * 2);
    _Float16*  qkvF16   = (_Float16*) carve((size_t)VROWS * 3 * VD * 2);
    float*     scoresF32= (float*)    carve((size_t)VN * VNH * VT * VT * 4);
    _Float16*  attF16   = (_Float16*) carve((size_t)VN * VNH * VT * VTP * 2);
    _Float16*  vT       = (_Float16*) carve((size_t)VN * VNH * VDH * VTP * 2);
    _Float16*  hF16     = (_Float16*) carve((size_t)VROWS * 4 * VD * 2);
    float*     tmpF32   = (float*)    carve((size_t)VROWS * 4 * VD * 4);
    _Float16*  wtF16    = (_Float16*) carve((size_t)4 * VD * VD * 2);   // max 3072x768
    _Float16*  featF16  = (_Float16*) carve((size_t)VN * VD * 2);
    _Float16*  headF16  = (_Float16*) carve((size_t)VN * 1000 * 2);
    _Float16*  fc1F16   = (_Float16*) carve((size_t)VN * 1000 * 2);

    const dim3 gblk(128);
    const dim3 tblk(32, 8);
    const long SA  = (long)VT * 3 * VD;         // qkv per-image stride
    const long SS  = (long)VT * VT;             // score tile size
    const long SAT = (long)VT * VTP;            // padded att tile
    const long SVT = (long)VDH * VTP;           // vT per-head stride

    // ---- preprocessing + patch embedding ----
    preprocess_kernel<<<VN * 16, 256, 0, stream>>>(x, noise, crop_off, jitter, img);
    patchify_kernel<<<(int)(((long)VN * 196 * VD + 255) / 256), 256, 0, stream>>>(img, patches);
    convert_f16_kernel<<<(int)(((long)VD * VD + 255) / 256), 256, 0, stream>>>(
        patch_w, wtF16, (long)VD * VD);                 // patch_w already [n][k]
    gemm_wmma_kernel<0, false, false><<<dim3(12, 25, 1), gblk, 0, stream>>>(
        patches, wtF16, patch_b, nullptr, tmpF32, nullptr,
        VN * 196, VD, VD, VD, VD, VD, 1, 0, 0, 0, 0, 0, 0);
    assemble_tokens_kernel<<<(int)(((long)VN * VT * VD + 255) / 256), 256, 0, stream>>>(
        tmpF32, cls_tok, pos_emb, tokF32);

    // ---- transformer layers ----
    for (int l = 0; l < VL; ++l) {
        ln_f16_kernel<<<VROWS, 256, 0, stream>>>(
            tokF32, VD, ln1_w + (long)l * VD, ln1_b + (long)l * VD, aF16, VD, VROWS);

        // qkv: W[768][2304] -> Wt[2304][768]
        transpose_f16_kernel<<<dim3(72, 24, 1), tblk, 0, stream>>>(
            qkv_w + (long)l * VD * 3 * VD, wtF16, VD, 3 * VD);
        gemm_wmma_kernel<0, false, true><<<dim3(36, 25, 1), gblk, 0, stream>>>(
            aF16, wtF16, qkv_b + (long)l * 3 * VD, nullptr, tmpF32, qkvF16,
            VROWS, 3 * VD, VD, VD, VD, 3 * VD, 1, 0, 0, 0, 0, 0, 0);

        // scores = Q @ K^T, batched (n,h): M=N=197, K=64
        gemm_wmma_kernel<0, false, false><<<dim3(4, 4, VN * VNH), gblk, 0, stream>>>(
            qkvF16, qkvF16 + VD, nullptr, nullptr, scoresF32, nullptr,
            VT, VT, VDH, 3 * VD, 3 * VD, VT, VNH,
            SA, VDH, SA, VDH, (long)VNH * SS, SS);

        softmax_f16_kernel<<<VN * VNH * VT, 256, 0, stream>>>(
            scoresF32, attF16, VT, VTP, 0.125f);
        repack_v_kernel<<<(int)(((long)VN * VNH * SVT + 255) / 256), 256, 0, stream>>>(
            qkvF16, vT);

        // o = att @ V: M=197, N=64, K=197; scatter to (N*T, D) at h*64
        gemm_wmma_kernel<0, false, true><<<dim3(1, 4, VN * VNH), gblk, 0, stream>>>(
            attF16, vT, nullptr, nullptr, tmpF32, aF16,
            VT, VDH, VT, VTP, VTP, VD, VNH,
            (long)VNH * SAT, SAT, (long)VNH * SVT, SVT, (long)VT * VD, VDH);

        // proj + residual
        transpose_f16_kernel<<<dim3(24, 24, 1), tblk, 0, stream>>>(
            proj_w + (long)l * VD * VD, wtF16, VD, VD);
        gemm_wmma_kernel<0, true, false><<<dim3(12, 25, 1), gblk, 0, stream>>>(
            aF16, wtF16, proj_b + (long)l * VD, tokF32, tokF32, nullptr,
            VROWS, VD, VD, VD, VD, VD, 1, 0, 0, 0, 0, 0, 0);

        ln_f16_kernel<<<VROWS, 256, 0, stream>>>(
            tokF32, VD, ln2_w + (long)l * VD, ln2_b + (long)l * VD, aF16, VD, VROWS);

        // mlp1 + GELU
        transpose_f16_kernel<<<dim3(96, 24, 1), tblk, 0, stream>>>(
            mlp_w1 + (long)l * VD * 4 * VD, wtF16, VD, 4 * VD);
        gemm_wmma_kernel<1, false, true><<<dim3(48, 25, 1), gblk, 0, stream>>>(
            aF16, wtF16, mlp_b1 + (long)l * 4 * VD, nullptr, tmpF32, hF16,
            VROWS, 4 * VD, VD, VD, VD, 4 * VD, 1, 0, 0, 0, 0, 0, 0);

        // mlp2 + residual
        transpose_f16_kernel<<<dim3(24, 96, 1), tblk, 0, stream>>>(
            mlp_w2 + (long)l * 4 * VD * VD, wtF16, 4 * VD, VD);
        gemm_wmma_kernel<0, true, false><<<dim3(12, 25, 1), gblk, 0, stream>>>(
            hF16, wtF16, mlp_b2 + (long)l * VD, tokF32, tokF32, nullptr,
            VROWS, VD, 4 * VD, 4 * VD, 4 * VD, VD, 1, 0, 0, 0, 0, 0, 0);
    }

    // ---- final LN on cls rows, head, fc1(relu), fc2 ----
    ln_f16_kernel<<<VN, 256, 0, stream>>>(
        tokF32, (long)VT * VD, lnf_w, lnf_b, featF16, VD, VN);

    transpose_f16_kernel<<<dim3(32, 24, 1), tblk, 0, stream>>>(head_w, wtF16, VD, 1000);
    gemm_wmma_kernel<0, false, true><<<dim3(16, 1, 1), gblk, 0, stream>>>(
        featF16, wtF16, head_b, nullptr, tmpF32, headF16,
        VN, 1000, VD, VD, VD, 1000, 1, 0, 0, 0, 0, 0, 0);

    transpose_f16_kernel<<<dim3(32, 32, 1), tblk, 0, stream>>>(fc1_w, wtF16, 1000, 1000);
    gemm_wmma_kernel<2, false, true><<<dim3(16, 1, 1), gblk, 0, stream>>>(
        headF16, wtF16, fc1_b, nullptr, tmpF32, fc1F16,
        VN, 1000, 1000, 1000, 1000, 1000, 1, 0, 0, 0, 0, 0, 0);

    transpose_f16_kernel<<<dim3(1, 32, 1), tblk, 0, stream>>>(fc2_w, wtF16, 1000, 32);
    gemm_wmma_kernel<0, false, false><<<dim3(1, 1, 1), gblk, 0, stream>>>(
        fc1F16, wtF16, fc2_b, nullptr, out, nullptr,
        VN, 32, 1000, 1000, 1000, 32, 1, 0, 0, 0, 0, 0, 0);

    target_kernel<<<1, 128, 0, stream>>>(noise, out + 256);
}
